// QwenMoE_40570261078426
// MI455X (gfx1250) — matrix operand order repaired and checked by reading, not compiled, not run on hardware
//
#include <hip/hip_runtime.h>
#include <hip/hip_bf16.h>
#include <math.h>

// ---------------------------------------------------------------------------
// Qwen MoE (top-4 of 60 experts + large shared expert) for MI455X / gfx1250.
// Memory-bound on streaming fp32 expert weights (~2.2 GB @ 23.3 TB/s => ~95us
// floor, intensity ~32 FLOP/B => need ~740 TFLOPS sustained matmul). All GEMMs
// run through v_wmma_f32_16x16x32_bf16 with on-the-fly fp32->bf16 conversion;
// routed tokens are gathered per expert (71 GFLOP instead of dense 531).
//
// GEMM inner loop: software-pipelined (unroll x2, two register sets), LDS
// double-buffered, ONE barrier per K-step, and the cvt+ds_store of the next
// tile is placed AFTER the WMMAs so global loads overlap matrix math (the
// loadcnt wait lands post-WMMA instead of being merged into the barrier).
// Weights transposed into LDS [n][k^swz] (stride 40 halfwords, XOR-8 swizzle:
// 2-way worst-case store conflicts, conflict-free b128 fragment reads).
// ---------------------------------------------------------------------------

typedef __bf16 bf16;
typedef __bf16 v16bf __attribute__((ext_vector_type(16)));
typedef __bf16 bfx8  __attribute__((ext_vector_type(8)));
typedef __bf16 bfx4  __attribute__((ext_vector_type(4)));
typedef float  v8f   __attribute__((ext_vector_type(8)));

#define T_TOK 512
#define HDIM  2048
#define NEXP  60
#define IDIM  1408
#define SDIM  5632
#define TOPK  4
#define MAXSLOTS 3072   // 512*4 assignments + <=15 pad per expert, rounded up

static __device__ __forceinline__ v16bf cat8(bfx8 lo, bfx8 hi) {
  return __builtin_shufflevector(lo, hi, 0,1,2,3,4,5,6,7,8,9,10,11,12,13,14,15);
}
// LDS k-swizzle: multiples of 8 only -> bfx8/bfx4 chunks stay contiguous.
static __device__ __forceinline__ int kswz(int n) { return 8 * ((n >> 4) & 3); }

// ---------------------------------------------------------------------------
// 1) Router: logits = x @ gate_w, softmax, top-4 -> route[T,E];
//    also sigmoid(x @ shared_gate_w) -> sig[T].
// ---------------------------------------------------------------------------
__global__ __launch_bounds__(64) void router_kernel(
    const float* __restrict__ x, const float* __restrict__ gw,
    const float* __restrict__ sgw, float* __restrict__ route,
    float* __restrict__ sig)
{
  __shared__ float xs[HDIM];
  __shared__ float lg[64];
  __shared__ float red[64];
  const int t = blockIdx.x, tid = threadIdx.x;
  for (int h = tid; h < HDIM; h += 64) xs[h] = x[(size_t)t * HDIM + h];
  __syncthreads();

  float acc = 0.f;
  if (tid < NEXP)
    for (int h = 0; h < HDIM; ++h) acc += xs[h] * gw[(size_t)h * NEXP + tid];
  lg[tid] = acc;

  float sg = 0.f;
  for (int h = tid; h < HDIM; h += 64) sg += xs[h] * sgw[h];
  red[tid] = sg;
  __syncthreads();

  if (tid == 0) {
    float s = 0.f;
    for (int i = 0; i < 64; ++i) s += red[i];
    sig[t] = 1.f / (1.f + __expf(-s));

    float mx = -1e30f;
    for (int e = 0; e < NEXP; ++e) mx = fmaxf(mx, lg[e]);
    float pr[NEXP]; float sum = 0.f;
    for (int e = 0; e < NEXP; ++e) { pr[e] = __expf(lg[e] - mx); sum += pr[e]; }
    const float inv = 1.f / sum;
    float outw[NEXP]; bool used[NEXP];
    for (int e = 0; e < NEXP; ++e) { outw[e] = 0.f; used[e] = false; }
    for (int k = 0; k < TOPK; ++k) {
      int bi = 0; float bv = -1.f;
      for (int e = 0; e < NEXP; ++e)
        if (!used[e] && pr[e] > bv) { bv = pr[e]; bi = e; }
      used[bi] = true; outw[bi] = pr[bi] * inv;
    }
    for (int e = 0; e < NEXP; ++e) route[(size_t)t * NEXP + e] = outw[e];
  }
}

// ---------------------------------------------------------------------------
// 2) Deterministic per-expert token lists, padded-to-16 prefix offsets.
// ---------------------------------------------------------------------------
__global__ __launch_bounds__(64) void build_lists_kernel(
    const float* __restrict__ route, int* __restrict__ offs,
    int* __restrict__ rowTok, float* __restrict__ rowScale)
{
  __shared__ int cs[NEXP];
  __shared__ int os[NEXP + 1];
  const int e = threadIdx.x;
  if (e < NEXP) {
    int c = 0;
    for (int t = 0; t < T_TOK; ++t) c += (route[(size_t)t * NEXP + e] > 0.f);
    cs[e] = c;
  }
  __syncthreads();
  if (e == 0) {
    int o = 0;
    for (int i = 0; i < NEXP; ++i) { os[i] = o; o += (cs[i] + 15) & ~15; }
    os[NEXP] = o;
    for (int i = 0; i <= NEXP; ++i) offs[i] = os[i];
  }
  __syncthreads();
  if (e < NEXP) {
    const int base = os[e];
    int j = 0;
    for (int t = 0; t < T_TOK; ++t) {
      const float w = route[(size_t)t * NEXP + e];
      if (w > 0.f) { rowTok[base + j] = t; rowScale[base + j] = w; ++j; }
    }
    const int pc = (cs[e] + 15) & ~15;
    for (; j < pc; ++j) { rowTok[base + j] = -1; rowScale[base + j] = 0.f; }
  }
}

// ---------------------------------------------------------------------------
// 3a) x fp32 -> bf16
// ---------------------------------------------------------------------------
__global__ __launch_bounds__(256) void cvt_x_kernel(
    const float* __restrict__ x, bf16* __restrict__ xb)
{
  const size_t i = ((size_t)blockIdx.x * 256 + threadIdx.x) * 8;
  const float4 a = *(const float4*)(x + i);
  const float4 b = *(const float4*)(x + i + 4);
  bfx8 v = { (bf16)a.x, (bf16)a.y, (bf16)a.z, (bf16)a.w,
             (bf16)b.x, (bf16)b.y, (bf16)b.z, (bf16)b.w };
  *(bfx8*)(xb + i) = v;
}

// ---------------------------------------------------------------------------
// 3b) Gather routed token rows (bf16), zero-fill pad slots.
// ---------------------------------------------------------------------------
__global__ __launch_bounds__(256) void gather_kernel(
    const bf16* __restrict__ xb, const int* __restrict__ offs,
    const int* __restrict__ rowTok, bf16* __restrict__ xg)
{
  const int e = blockIdx.x, mt = blockIdx.y, tid = threadIdx.x;
  const int o0 = offs[e], padded = offs[e + 1] - o0;
  if (mt * 16 >= padded) return;
  const int base = o0 + mt * 16;
  for (int s = 0; s < 16; ++s) {
    const int slot = base + s, tok = rowTok[slot];
    bfx8 v = {};
    if (tok >= 0) v = *(const bfx8*)(xb + (size_t)tok * HDIM + tid * 8);
    *(bfx8*)(xg + (size_t)slot * HDIM + tid * 8) = v;
  }
}

// ---------------------------------------------------------------------------
// WMMA fragment layouts (CDNA5 ISA §7.12.2), wave32:
//  A (16x32 bf16): lane l -> row l%16, K = kb..kb+15 (code's order, same as B),
//                  kb = (l<16)?0:16.
//  B (32x16 bf16): lane l -> col l%16, K = kb..kb+15, kb = (l<16)?0:16 (code's order, not the ISA table's).
//  C/D (16x16 f32): lane l, vgpr r -> M = r + 8*(l>=16), N = l%16.
// Staging: wave w owns k-rows 4w..4w+3 of the 32xN k-tile, lane owns 4
// consecutive n columns -> coalesced fp32 loads; transpose-store as bfx4
// (ds_store_b64) into lds[n][k^swz(n)], row stride 40 halfwords.
// ---------------------------------------------------------------------------

static __device__ __forceinline__ void load_w2(
    const float* __restrict__ W1, const float* __restrict__ W3, int kk, int N,
    float4 (&r1)[4], float4 (&r3)[4])
{
  #pragma unroll
  for (int j = 0; j < 4; ++j) {
    r1[j] = *(const float4*)(W1 + (size_t)(kk + j) * N);
    r3[j] = *(const float4*)(W3 + (size_t)(kk + j) * N);
  }
}
static __device__ __forceinline__ void load_a(
    const bf16* __restrict__ A, int kk, int akb, bfx8& alo, bfx8& ahi)
{
  alo = *(const bfx8*)(A + kk + akb);
  ahi = *(const bfx8*)(A + kk + akb + 8);
}
static __device__ __forceinline__ void stash2(
    bf16 (*buf)[128][40], int lane, int wv,
    const float4 (&r1)[4], const float4 (&r3)[4])
{
  #pragma unroll
  for (int q = 0; q < 4; ++q) {
    const int n  = 4 * lane + q;
    const int kp = (4 * wv) ^ kswz(n);
    const bfx4 p1 = { (bf16)((const float*)&r1[0])[q], (bf16)((const float*)&r1[1])[q],
                      (bf16)((const float*)&r1[2])[q], (bf16)((const float*)&r1[3])[q] };
    const bfx4 p3 = { (bf16)((const float*)&r3[0])[q], (bf16)((const float*)&r3[1])[q],
                      (bf16)((const float*)&r3[2])[q], (bf16)((const float*)&r3[3])[q] };
    *(bfx4*)&buf[0][n][kp] = p1;
    *(bfx4*)&buf[1][n][kp] = p3;
  }
}
static __device__ __forceinline__ void stash1(
    bf16 (*buf)[40], int lane, int wv, const float4 (&r)[4])
{
  #pragma unroll
  for (int q = 0; q < 4; ++q) {
    const int n  = 4 * lane + q;
    const int kp = (4 * wv) ^ kswz(n);
    const bfx4 p = { (bf16)((const float*)&r[0])[q], (bf16)((const float*)&r[1])[q],
                     (bf16)((const float*)&r[2])[q], (bf16)((const float*)&r[3])[q] };
    *(bfx4*)&buf[n][kp] = p;
  }
}

// 4) Up-projection dual GEMM: act = silu(A @ W1) * (A @ W3), bf16 out.
__global__ __launch_bounds__(256) void up_kernel(
    const bf16* __restrict__ Abase, const float* __restrict__ W1g,
    const float* __restrict__ W3g, const int* __restrict__ offs,
    int fixedRows, int K, int N, unsigned long long wStride,
    bf16* __restrict__ act)
{
  const int e = blockIdx.x, nt = blockIdx.y, mt = blockIdx.z;
  int rowBase, padded;
  if (offs) { const int o0 = offs[e]; padded = offs[e + 1] - o0; rowBase = o0 + mt * 16; }
  else      { padded = fixedRows; rowBase = mt * 16; }
  if (mt * 16 >= padded) return;

  const int tid  = threadIdx.x;
  const int lane = tid & 31;
  const int wv   = tid >> 5;
  const int mrow = lane & 15;
  const int akb  = (lane < 16) ? 0 : 16;
  const int bkb  = (lane < 16) ? 0 : 16;
  const int ncol = wv * 16 + (lane & 15);
  const int rswz = kswz(ncol);

  const float* W1 = W1g + (unsigned long long)e * wStride
                  + (size_t)(4 * wv) * N + (size_t)nt * 128 + 4 * lane;
  const float* W3 = W3g + (unsigned long long)e * wStride
                  + (size_t)(4 * wv) * N + (size_t)nt * 128 + 4 * lane;
  const bf16*  A  = Abase + (size_t)(rowBase + mrow) * K;

  __shared__ bf16 lds[2][2][128][40];   // [buf][mat][n][k'], 40960 B

  float4 r1a[4], r3a[4], r1b[4], r3b[4];
  bfx8 aloA, ahiA, aloB, ahiB;
  load_w2(W1, W3, 0, N, r1a, r3a);
  load_a(A, 0, akb, aloA, ahiA);
  stash2(lds[0], lane, wv, r1a, r3a);

  v8f acc1 = {}; v8f acc3 = {};
  // K is a multiple of 64 for all call sites (2048 at both).
  for (int k0 = 0; k0 < K; k0 += 64) {
    { // stage 0: compute tile k0 from lds[0]; load k0+32 -> B; stash -> lds[1]
      __syncthreads();
      load_w2(W1, W3, k0 + 32, N, r1b, r3b);
      load_a(A, k0 + 32, akb, aloB, ahiB);
      const int pk = (k0 + 64 < K) ? k0 + 64 : k0 + 32;
      __builtin_prefetch(W1 + (size_t)pk * N, 0, 1);
      __builtin_prefetch(W3 + (size_t)pk * N, 0, 1);
      const v16bf a  = cat8(aloA, ahiA);
      const v16bf b1 = cat8(*(const bfx8*)&lds[0][0][ncol][bkb ^ rswz],
                            *(const bfx8*)&lds[0][0][ncol][(bkb + 8) ^ rswz]);
      const v16bf b3 = cat8(*(const bfx8*)&lds[0][1][ncol][bkb ^ rswz],
                            *(const bfx8*)&lds[0][1][ncol][(bkb + 8) ^ rswz]);
      acc1 = __builtin_amdgcn_wmma_f32_16x16x32_bf16(false, a, false, b1, (short)0, acc1, false, false);
      acc3 = __builtin_amdgcn_wmma_f32_16x16x32_bf16(false, a, false, b3, (short)0, acc3, false, false);
      stash2(lds[1], lane, wv, r1b, r3b);   // loadcnt wait lands here, post-WMMA
    }
    { // stage 1: compute tile k0+32 from lds[1]; load k0+64 -> A; stash -> lds[0]
      __syncthreads();
      const int kn = (k0 + 64 < K) ? k0 + 64 : k0 + 32;  // last iter: dead reload
      load_w2(W1, W3, kn, N, r1a, r3a);
      load_a(A, kn, akb, aloA, ahiA);
      const int pk = (kn + 32 < K) ? kn + 32 : kn;
      __builtin_prefetch(W1 + (size_t)pk * N, 0, 1);
      __builtin_prefetch(W3 + (size_t)pk * N, 0, 1);
      const v16bf a  = cat8(aloB, ahiB);
      const v16bf b1 = cat8(*(const bfx8*)&lds[1][0][ncol][bkb ^ rswz],
                            *(const bfx8*)&lds[1][0][ncol][(bkb + 8) ^ rswz]);
      const v16bf b3 = cat8(*(const bfx8*)&lds[1][1][ncol][bkb ^ rswz],
                            *(const bfx8*)&lds[1][1][ncol][(bkb + 8) ^ rswz]);
      acc1 = __builtin_amdgcn_wmma_f32_16x16x32_bf16(false, a, false, b1, (short)0, acc1, false, false);
      acc3 = __builtin_amdgcn_wmma_f32_16x16x32_bf16(false, a, false, b3, (short)0, acc3, false, false);
      stash2(lds[0], lane, wv, r1a, r3a);
    }
  }

  #pragma unroll
  for (int r = 0; r < 8; ++r) {
    const int m = r + ((lane < 16) ? 0 : 8);
    const float g = acc1[r], u = acc3[r];
    const float s = g * __builtin_amdgcn_rcpf(1.f + __expf(-g));   // fast silu
    act[(size_t)(rowBase + m) * N + (size_t)nt * 128 + ncol] = (bf16)(s * u);
  }
}

// ---------------------------------------------------------------------------
// 5) Down-projection GEMM: out[tok] (+)= scale[slot] * (act @ W2).
//    atomicFlag=0: plain store (shared expert, covers every output once).
//    atomicFlag=1: atomicAdd (MoE accumulation on top).
// ---------------------------------------------------------------------------
__global__ __launch_bounds__(256) void down_kernel(
    const bf16* __restrict__ Abase, const float* __restrict__ W2g,
    const int* __restrict__ offs, int fixedRows,
    const int* __restrict__ rowTok, const float* __restrict__ rowScale,
    int K, unsigned long long wStride, float* __restrict__ out, int atomicFlag)
{
  const int N = HDIM;
  const int e = blockIdx.x, nt = blockIdx.y, mt = blockIdx.z;
  int rowBase, padded;
  if (offs) { const int o0 = offs[e]; padded = offs[e + 1] - o0; rowBase = o0 + mt * 16; }
  else      { padded = fixedRows; rowBase = mt * 16; }
  if (mt * 16 >= padded) return;

  const int tid  = threadIdx.x;
  const int lane = tid & 31;
  const int wv   = tid >> 5;
  const int mrow = lane & 15;
  const int akb  = (lane < 16) ? 0 : 16;
  const int bkb  = (lane < 16) ? 0 : 16;
  const int ncol = wv * 16 + (lane & 15);
  const int rswz = kswz(ncol);

  const float* W2 = W2g + (unsigned long long)e * wStride
                  + (size_t)(4 * wv) * N + (size_t)nt * 128 + 4 * lane;
  const bf16*  A  = Abase + (size_t)(rowBase + mrow) * K;

  __shared__ bf16 lds[2][128][40];      // [buf][n][k'], 20480 B

  float4 rwa[4], rwb[4];
  bfx8 aloA, ahiA, aloB, ahiB;
  #pragma unroll
  for (int j = 0; j < 4; ++j) rwa[j] = *(const float4*)(W2 + (size_t)j * N);
  load_a(A, 0, akb, aloA, ahiA);
  stash1(lds[0], lane, wv, rwa);

  v8f acc = {};
  // K is a multiple of 64 for all call sites (1408 / 5632).
  for (int k0 = 0; k0 < K; k0 += 64) {
    {
      __syncthreads();
      #pragma unroll
      for (int j = 0; j < 4; ++j) rwb[j] = *(const float4*)(W2 + (size_t)(k0 + 32 + j) * N);
      load_a(A, k0 + 32, akb, aloB, ahiB);
      const int pk = (k0 + 64 < K) ? k0 + 64 : k0 + 32;
      __builtin_prefetch(W2 + (size_t)pk * N, 0, 1);
      const v16bf a = cat8(aloA, ahiA);
      const v16bf b = cat8(*(const bfx8*)&lds[0][ncol][bkb ^ rswz],
                           *(const bfx8*)&lds[0][ncol][(bkb + 8) ^ rswz]);
      acc = __builtin_amdgcn_wmma_f32_16x16x32_bf16(false, a, false, b, (short)0, acc, false, false);
      stash1(lds[1], lane, wv, rwb);
    }
    {
      __syncthreads();
      const int kn = (k0 + 64 < K) ? k0 + 64 : k0 + 32;
      #pragma unroll
      for (int j = 0; j < 4; ++j) rwa[j] = *(const float4*)(W2 + (size_t)(kn + j) * N);
      load_a(A, kn, akb, aloA, ahiA);
      const int pk = (kn + 32 < K) ? kn + 32 : kn;
      __builtin_prefetch(W2 + (size_t)pk * N, 0, 1);
      const v16bf a = cat8(aloB, ahiB);
      const v16bf b = cat8(*(const bfx8*)&lds[1][ncol][bkb ^ rswz],
                           *(const bfx8*)&lds[1][ncol][(bkb + 8) ^ rswz]);
      acc = __builtin_amdgcn_wmma_f32_16x16x32_bf16(false, a, false, b, (short)0, acc, false, false);
      stash1(lds[0], lane, wv, rwa);
    }
  }

  #pragma unroll
  for (int r = 0; r < 8; ++r) {
    const int m = r + ((lane < 16) ? 0 : 8);
    const int slot = rowBase + m;
    const int tok = rowTok ? rowTok[slot] : slot;
    if (tok < 0) continue;
    const float v = acc[r] * rowScale[slot];
    float* dst = out + (size_t)tok * N + (size_t)nt * 128 + ncol;
    if (atomicFlag) atomicAdd(dst, v);
    else            *dst = v;
  }
}

// ---------------------------------------------------------------------------
// Launch. Workspace (~30 MB): route, offsets, token lists, bf16 x, gathered
// bf16 rows, bf16 intermediate activations for MoE + shared expert.
// ---------------------------------------------------------------------------
extern "C" void kernel_launch(void* const* d_in, const int* in_sizes, int n_in,
                              void* d_out, int out_size, void* d_ws, size_t ws_size,
                              hipStream_t stream) {
  (void)in_sizes; (void)n_in; (void)out_size; (void)ws_size;
  const float* x   = (const float*)d_in[0];
  const float* gw  = (const float*)d_in[1];
  const float* w1  = (const float*)d_in[2];
  const float* w3  = (const float*)d_in[3];
  const float* w2  = (const float*)d_in[4];
  const float* sw1 = (const float*)d_in[5];
  const float* sw3 = (const float*)d_in[6];
  const float* sw2 = (const float*)d_in[7];
  const float* sgw = (const float*)d_in[8];
  float* out = (float*)d_out;

  uintptr_t p = (uintptr_t)d_ws;
  auto carve = [&](size_t bytes) -> void* {
    void* r = (void*)p; p += (bytes + 255) & ~(size_t)255; return r;
  };
  float* route    = (float*)carve((size_t)T_TOK * NEXP * 4);
  int*   offs     = (int*)  carve((NEXP + 1) * 4);
  int*   rowTok   = (int*)  carve(MAXSLOTS * 4);
  float* rowScale = (float*)carve(MAXSLOTS * 4);
  float* sig      = (float*)carve(T_TOK * 4);
  bf16*  xb       = (bf16*) carve((size_t)T_TOK * HDIM * 2);
  bf16*  xg       = (bf16*) carve((size_t)MAXSLOTS * HDIM * 2);
  bf16*  actm     = (bf16*) carve((size_t)MAXSLOTS * IDIM * 2);
  bf16*  acts     = (bf16*) carve((size_t)T_TOK * SDIM * 2);

  router_kernel<<<T_TOK, 64, 0, stream>>>(x, gw, sgw, route, sig);
  build_lists_kernel<<<1, 64, 0, stream>>>(route, offs, rowTok, rowScale);
  cvt_x_kernel<<<(T_TOK * HDIM) / (256 * 8), 256, 0, stream>>>(x, xb);
  gather_kernel<<<dim3(NEXP, 32), 256, 0, stream>>>(xb, offs, rowTok, xg);

  // MoE up-projection (gathered rows) and shared-expert up-projection.
  up_kernel<<<dim3(NEXP, IDIM / 128, 32), 256, 0, stream>>>(
      xg, w1, w3, offs, 0, HDIM, IDIM, (unsigned long long)HDIM * IDIM, actm);
  up_kernel<<<dim3(1, SDIM / 128, T_TOK / 16), 256, 0, stream>>>(
      xb, sw1, sw3, nullptr, T_TOK, HDIM, SDIM, 0ULL, acts);

  // Shared expert down-proj writes out (sigmoid-gated), then MoE atomically
  // accumulates route-scaled expert contributions on top.
  down_kernel<<<dim3(1, HDIM / 128, T_TOK / 16), 256, 0, stream>>>(
      acts, sw2, nullptr, T_TOK, nullptr, sig, SDIM, 0ULL, out, 0);
  down_kernel<<<dim3(NEXP, HDIM / 128, 32), 256, 0, stream>>>(
      actm, w2, offs, 0, rowTok, rowScale, IDIM,
      (unsigned long long)IDIM * HDIM, out, 1);
}
